// GATModel_15685220565801
// MI455X (gfx1250) — matrix operand order repaired and checked by reading, not compiled, not run on hardware
//
#include <hip/hip_runtime.h>
#include <hip/hip_bf16.h>

typedef __attribute__((ext_vector_type(16))) _Float16 v16h;
typedef __attribute__((ext_vector_type(8)))  _Float16 v8h;
typedef __attribute__((ext_vector_type(8)))  float    v8f;

#define NN   40      // nodes per graph
#define HIDN 128     // hidden dim
#define XROW 240

// ---------- WMMA fragment loaders (16-bit A 16x32 / B 32x16 layouts, ISA 7.12.2) ----------
__device__ __forceinline__ v16h load_a16(const _Float16* __restrict__ row, int kb, int hi) {
  v16h a;
#pragma unroll
  for (int t = 0; t < 8; ++t) a[t] = row[kb + hi * 8 + t];          // K = kb..kb+7 / kb+8..15
#pragma unroll
  for (int t = 0; t < 8; ++t) a[8 + t] = row[kb + 16 + hi * 8 + t]; // K = kb+16.. / kb+24..
  return a;
}
__device__ __forceinline__ v16h load_b16(const _Float16* __restrict__ row, int kb2) {
  v16h b;
  const int kb = kb2 & ~31, h = (kb2 >> 4) & 1;
#pragma unroll
  for (int t = 0; t < 8; ++t) b[t] = row[kb + h * 8 + t];           // K = kb..kb+7 / kb+8..15
#pragma unroll
  for (int t = 0; t < 8; ++t) b[8 + t] = row[kb + 16 + h * 8 + t];  // K = kb+16.. / kb+24..
  return b;
}
__device__ __forceinline__ float sigm(float x) { return 1.0f / (1.0f + __expf(-x)); }

// ---------- prep: transpose proc_W (3x128x128 f32, K x N) -> f16 WT[l][n][k] ----------
__global__ void gat_prep_wt(const float* __restrict__ procW, _Float16* __restrict__ wtg) {
  int idx = blockIdx.x * 256 + threadIdx.x;
  if (idx < 3 * HIDN * HIDN) {
    int l = idx >> 14, r = idx & 16383;
    int n = r >> 7, k = r & 127;
    wtg[idx] = (_Float16)procW[l * 16384 + k * HIDN + n];
  }
}

// ---------- fused per-graph GAT forward ----------
__global__ __launch_bounds__(256) void gat_fused(
    const float* __restrict__ x,
    const float* __restrict__ encW, const float* __restrict__ encAs,
    const float* __restrict__ encAd, const float* __restrict__ encB,
    const _Float16* __restrict__ wtg,  // [3][128][128] f16, N-major (pre-transposed)
    const float* __restrict__ procAs, const float* __restrict__ procAd,
    const float* __restrict__ procB,
    const float* __restrict__ decW, const float* __restrict__ decAs,
    const float* __restrict__ decAd, const float* __restrict__ decB,
    float* __restrict__ out) {
  __shared__ _Float16 sh_h16[48 * HIDN];   // node features h, f16, rows 40..47 zero
  __shared__ _Float16 sh_gT[HIDN * 72];    // projected features transposed gT[n][k], k-pad zero
  __shared__ __align__(16) unsigned char arena[32768]; // WT (proj) OR softmax/decoder scratch

  _Float16* WT     = (_Float16*)arena;                // 32768 B, live copy->proj only
  float*    s_s    = (float*)(arena + 0);             // 160 B
  float*    s_d    = (float*)(arena + 256);           // 160 B
  float*    s_feat = (float*)(arena + 512);           // 480 B
  _Float16* s_al   = (_Float16*)(arena + 1024);       // alpha f16 [48][64] = 6144 B
  float*    s_gdec = (float*)(arena + 7680);          // 480 B
  float*    s_adec = (float*)(arena + 8192);          // 6400 B
  float*    s_outn = (float*)(arena + 16384);         // 480 B

  const int tid  = threadIdx.x;
  const int lane = tid & 31, wave = tid >> 5;
  const int hi = lane >> 4, ln = lane & 15;
  const int b = blockIdx.x;

  // zero pads once (never overwritten afterwards)
  for (int i = tid; i < 48 * HIDN; i += 256) sh_h16[i] = (_Float16)0.0f;
  for (int i = tid; i < HIDN * 72; i += 256) sh_gT[i] = (_Float16)0.0f;
  for (int i = tid; i < NN * 3; i += 256) s_feat[i] = x[b * XROW + 120 + i];
  __syncthreads();

  // ================= encoder projection (40x3 @ 3x128, scalar) =================
  for (int o = tid; o < NN * HIDN; o += 256) {
    int m = o >> 7, n = o & 127;
    float v = s_feat[m * 3 + 0] * encW[0 * HIDN + n] +
              s_feat[m * 3 + 1] * encW[1 * HIDN + n] +
              s_feat[m * 3 + 2] * encW[2 * HIDN + n];
    sh_gT[n * 72 + m] = (_Float16)v;
  }
  __syncthreads();

  for (int layer = 0; layer < 4; ++layer) {  // 0 = encoder, 1..3 = processors
    const float* as = (layer == 0) ? encAs : procAs + (layer - 1) * HIDN;
    const float* ad = (layer == 0) ? encAd : procAd + (layer - 1) * HIDN;
    const float* bs = (layer == 0) ? encB  : procB  + (layer - 1) * HIDN;

    if (layer > 0) {
      // ---- processor projection: copy WT f16 (contiguous), then WMMA g = h @ W ----
      const uint4* src = (const uint4*)(wtg + (layer - 1) * HIDN * HIDN);
      uint4* dst = (uint4*)WT;
      for (int i = tid; i < HIDN * HIDN / 8; i += 256) dst[i] = src[i];
      __syncthreads();
#pragma unroll
      for (int t = 0; t < 3; ++t) {
        int tile = wave + t * 8;              // 24 tiles: 3 M-tiles x 8 N-tiles
        int mt = tile >> 3, nt = tile & 7;
        v8f acc = {};
        const _Float16* arow = &sh_h16[(mt * 16 + ln) * HIDN];
        const _Float16* brow = &WT[(nt * 16 + ln) * HIDN];
#pragma unroll
        for (int kt = 0; kt < 4; ++kt) {
          v16h a  = load_a16(arow, kt * 32, hi);
          v16h bb = load_b16(brow, kt * 32 + hi * 16);
          acc = __builtin_amdgcn_wmma_f32_16x16x32_f16(false, a, false, bb,
                                                       (short)0, acc, false, false);
        }
        // packed C-store: 8 consecutive m at fixed n -> one b128; guard uniform over r
        if (mt < 2 || hi == 0) {
          int n = nt * 16 + ln;
          v8h hv;
#pragma unroll
          for (int r = 0; r < 8; ++r) hv[r] = (_Float16)acc[r];
          *(v8h*)&sh_gT[n * 72 + mt * 16 + hi * 8] = hv;
        }
      }
      __syncthreads();
    }

    // ---- per-node logits s,d; re-zero alpha padding (arena was WT) ----
    if (tid < NN) {
      float ss = 0.f, dd = 0.f;
      for (int n = 0; n < HIDN; ++n) {
        float g = (float)sh_gT[n * 72 + tid];
        ss += g * as[n];
        dd += g * ad[n];
      }
      s_s[tid] = ss; s_d[tid] = dd;
    }
    for (int idx = tid; idx < 48 * 64; idx += 256) {
      int r = idx >> 6, c = idx & 63;
      if (r >= NN || c >= NN) s_al[idx] = (_Float16)0.0f;  // zero pad -> no NaN into WMMA
    }
    __syncthreads();

    // ---- softmax over sources (leaky-relu logits), 3-pass, one row per thread ----
    if (tid < NN) {
      float di = s_d[tid], mx = -3.4e38f;
      for (int j = 0; j < NN; ++j) {
        float v = di + s_s[j]; v = (v > 0.f) ? v : 0.2f * v;
        mx = fmaxf(mx, v);
      }
      float sum = 0.f;
      for (int j = 0; j < NN; ++j) {
        float v = di + s_s[j]; v = (v > 0.f) ? v : 0.2f * v;
        sum += __expf(v - mx);
      }
      float inv = 1.0f / sum;
      for (int j = 0; j < NN; ++j) {
        float v = di + s_s[j]; v = (v > 0.f) ? v : 0.2f * v;
        s_al[tid * 64 + j] = (_Float16)(__expf(v - mx) * inv);
      }
    }
    __syncthreads();

    // ---- attention, transposed: h_new^T = g^T @ alpha^T  (A = gT rows, B = alpha rows) ----
    // D[mrow=feature][ncol=node]; each lane owns 8 consecutive features of one node.
    {
      const int ft  = wave;                 // feature tile 0..7 (M = 128)
      const int nf0 = ft * 16 + hi * 8;     // first of this lane's 8 features
      float4 bv0 = ((const float4*)(bs + nf0))[0];
      float4 bv1 = ((const float4*)(bs + nf0))[1];
      float bias8[8] = {bv0.x, bv0.y, bv0.z, bv0.w, bv1.x, bv1.y, bv1.z, bv1.w};
      const _Float16* arow = &sh_gT[(ft * 16 + ln) * 72];
#pragma unroll
      for (int ntl = 0; ntl < 3; ++ntl) {   // node tiles (N = 48)
        v8f acc = {};
        const _Float16* brow = &s_al[(ntl * 16 + ln) * 64];
#pragma unroll
        for (int kt = 0; kt < 2; ++kt) {    // K = 64 (nodes, zero-padded)
          v16h a  = load_a16(arow, kt * 32, hi);
          v16h bb = load_b16(brow, kt * 32 + hi * 16);
          acc = __builtin_amdgcn_wmma_f32_16x16x32_f16(false, a, false, bb,
                                                       (short)0, acc, false, false);
        }
        int m = ntl * 16 + ln;              // node index (uniform over r)
        if (m < NN) {
          _Float16* hrow = &sh_h16[m * HIDN + nf0];
          v8h hv;
          if (layer > 0) {
            v8h hold = *(const v8h*)hrow;   // residual, one b128 load
#pragma unroll
            for (int r = 0; r < 8; ++r)
              hv[r] = (_Float16)((float)hold[r] + sigm(acc[r] + bias8[r]));
          } else {
#pragma unroll
            for (int r = 0; r < 8; ++r)
              hv[r] = (_Float16)sigm(acc[r] + bias8[r]);
          }
          *(v8h*)hrow = hv;                 // one b128 store
        }
      }
    }
    __syncthreads();
  }

  // ================= decoder GAT (128 -> 3), scalar, then node-sum =================
  if (tid < NN * 3) {
    int m = tid / 3, c = tid % 3;
    float v = 0.f;
    for (int k = 0; k < HIDN; ++k) v += (float)sh_h16[m * HIDN + k] * decW[k * 3 + c];
    s_gdec[tid] = v;
  }
  __syncthreads();
  if (tid < NN) {
    float ss = 0.f, dd = 0.f;
    for (int c = 0; c < 3; ++c) {
      ss += s_gdec[tid * 3 + c] * decAs[c];
      dd += s_gdec[tid * 3 + c] * decAd[c];
    }
    s_s[tid] = ss; s_d[tid] = dd;
  }
  __syncthreads();
  if (tid < NN) {
    float di = s_d[tid], mx = -3.4e38f;
    for (int j = 0; j < NN; ++j) {
      float v = di + s_s[j]; v = (v > 0.f) ? v : 0.2f * v;
      mx = fmaxf(mx, v);
    }
    float sum = 0.f;
    for (int j = 0; j < NN; ++j) {
      float v = di + s_s[j]; v = (v > 0.f) ? v : 0.2f * v;
      sum += __expf(v - mx);
    }
    float inv = 1.0f / sum;
    for (int j = 0; j < NN; ++j) {
      float v = di + s_s[j]; v = (v > 0.f) ? v : 0.2f * v;
      s_adec[tid * 40 + j] = __expf(v - mx) * inv;
    }
  }
  __syncthreads();
  if (tid < NN * 3) {
    int i = tid / 3, c = tid % 3;
    float v = 0.f;
    for (int j = 0; j < NN; ++j) v += s_adec[i * 40 + j] * s_gdec[j * 3 + c];
    s_outn[tid] = v;
  }
  __syncthreads();
  if (tid < 3) {
    float sum = 0.f;
    for (int i = 0; i < NN; ++i) sum += s_outn[i * 3 + tid];
    out[b * 3 + tid] = sum + 40.0f * decB[tid];  // bias summed over 40 nodes
  }
}

extern "C" void kernel_launch(void* const* d_in, const int* in_sizes, int n_in,
                              void* d_out, int out_size, void* d_ws, size_t ws_size,
                              hipStream_t stream) {
  const float* x      = (const float*)d_in[0];
  const float* encW   = (const float*)d_in[1];
  const float* encAs  = (const float*)d_in[2];
  const float* encAd  = (const float*)d_in[3];
  const float* encB   = (const float*)d_in[4];
  const float* procW  = (const float*)d_in[5];
  const float* procAs = (const float*)d_in[6];
  const float* procAd = (const float*)d_in[7];
  const float* procB  = (const float*)d_in[8];
  const float* decW   = (const float*)d_in[9];
  const float* decAs  = (const float*)d_in[10];
  const float* decAd  = (const float*)d_in[11];
  const float* decB   = (const float*)d_in[12];
  float* out = (float*)d_out;

  const int B = in_sizes[0] / XROW;           // 16384
  _Float16* wtg = (_Float16*)d_ws;            // 3*128*128 f16 = 96 KB scratch

  gat_prep_wt<<<(3 * HIDN * HIDN + 255) / 256, 256, 0, stream>>>(procW, wtg);
  gat_fused<<<B, 256, 0, stream>>>(x, encW, encAs, encAd, encB, wtg,
                                   procAs, procAd, procB,
                                   decW, decAs, decAd, decB, out);
}
